// SapBERTembeddedSOM_26525718020323
// MI455X (gfx1250) — matrix operand order repaired and checked by reading, not compiled, not run on hardware
//
#include <hip/hip_runtime.h>
#include <hip/hip_bf16.h>

// MI455X (gfx1250, wave32) SOM step.
// Both big GEMMs run on v_wmma_f32_16x16x32_bf16 with error-compensated
// split-bf16 operands (hi+lo, 3 products) and f32 accumulators -> ~f32 accuracy
// at bf16 matrix-core rates. All split operands fit in the 192MB L2, so fragments
// are loaded directly as per-lane 16B chunks (global_load_b128) and L2 does the
// blocking. BMU chunks 4 col-tiles per accumulator group to amortize A-fragment
// loads (1.67 loads/WMMA), and prefetches the next chunk (global_prefetch_b8).
// The smoothing kernel builds exp() W tiles in LDS once per k-step, shares them
// across 8 waves, and fuses the W row-sum (so grid_dists is read exactly once).

typedef __attribute__((ext_vector_type(16))) __bf16 v16bf;
typedef __attribute__((ext_vector_type(8)))  __bf16 v8bf;
typedef __attribute__((ext_vector_type(8)))  float  v8f;

#define NBATCH 8192
#define COMPS  4096
#define DIM    768

__device__ __forceinline__ v8f wmma_bf16(v16bf a, v16bf b, v8f c) {
  return __builtin_amdgcn_wmma_f32_16x16x32_bf16(false, a, false, b, (short)0, c,
                                                 false, false);
}

// ---------------------------------------------------------------- zero scratch
__global__ __launch_bounds__(256) void zero_kernel(float* freq, float* sums, float* out) {
  size_t idx = (size_t)blockIdx.x * 256 + threadIdx.x;
  if (idx == 0) out[0] = 0.0f;
  if (idx < COMPS) freq[idx] = 0.0f;
  size_t n = (size_t)COMPS * DIM;
  size_t stride = (size_t)gridDim.x * 256;
  for (size_t k = idx; k < n; k += stride) sums[k] = 0.0f;
}

// -------------------------------------------------- f32 -> (bf16 hi, bf16 lo)
__global__ __launch_bounds__(256) void split_kernel(const float* __restrict__ x,
                                                    __bf16* __restrict__ hi,
                                                    __bf16* __restrict__ lo, size_t n) {
  size_t idx = (size_t)blockIdx.x * 256 + threadIdx.x;
  if (idx < n) {
    float v = x[idx];
    __bf16 h = (__bf16)v;
    hi[idx] = h;
    lo[idx] = (__bf16)(v - (float)h);
  }
}

// ------------------------------------------------------- ||c_j||^2 per comp j
__global__ __launch_bounds__(256) void cnorm_kernel(const float* __restrict__ C,
                                                    float* __restrict__ cnorm) {
  __shared__ float red[256];
  const int j = blockIdx.x;
  float s = 0.0f;
  for (int k = threadIdx.x; k < DIM; k += 256) {
    float v = C[(size_t)j * DIM + k];
    s += v * v;
  }
  red[threadIdx.x] = s;
  __syncthreads();
  for (int m = 128; m > 0; m >>= 1) {
    if ((int)threadIdx.x < m) red[threadIdx.x] += red[threadIdx.x + m];
    __syncthreads();
  }
  if (threadIdx.x == 0) cnorm[j] = red[0];
}

// ---------------------------------------------- BMU: argmin_j (|c|^2 - 2 e.c)
// block = 16 E-rows x 8 waves; wave w scans cols [w*512, w*512+512) in chunks
// of 4 col-tiles so each A fragment feeds 12 WMMAs.
__global__ __launch_bounds__(256) void bmu_kernel(const __bf16* __restrict__ Ehi,
                                                  const __bf16* __restrict__ Elo,
                                                  const __bf16* __restrict__ Chi,
                                                  const __bf16* __restrict__ Clo,
                                                  const float* __restrict__ cnorm,
                                                  int* __restrict__ minidx) {
  __shared__ float sval[8][16];
  __shared__ int   sidx[8][16];
  const int lane = threadIdx.x & 31;
  const int wave = threadIdx.x >> 5;
  const int l16  = lane & 15;
  const int hf   = lane >> 4;
  const int row0 = blockIdx.x * 16;

  const __bf16* arhi = Ehi + (size_t)(row0 + l16) * DIM;
  const __bf16* arlo = Elo + (size_t)(row0 + l16) * DIM;

  float minv[8];
  int   mini[8];
#pragma unroll
  for (int r = 0; r < 8; ++r) { minv[r] = 3.4e38f; mini[r] = 0x7fffffff; }

  for (int cc = 0; cc < 8; ++cc) {
    const int jbase = wave * 512 + cc * 64;
    const __bf16* brhi[4];
    const __bf16* brlo[4];
#pragma unroll
    for (int c = 0; c < 4; ++c) {
      brhi[c] = Chi + (size_t)(jbase + c * 16 + l16) * DIM;
      brlo[c] = Clo + (size_t)(jbase + c * 16 + l16) * DIM;
    }
    if (cc < 7) {  // warm next chunk's rows (global_prefetch_b8)
#pragma unroll
      for (int c = 0; c < 4; ++c) {
        __builtin_prefetch(brhi[c] + 64 * DIM, 0, 3);
        __builtin_prefetch(brlo[c] + 64 * DIM, 0, 3);
      }
    }
    v8f acc[4] = {};
#pragma unroll 2
    for (int kb = 0; kb < 24; ++kb) {
      const int k0 = kb * 32;
      const int ak = k0 + hf * 16;   // A frag: K 0..15 / 16..31, contiguous
      const int bk = k0 + hf * 16;   // B frag: K 0..15 / 16..31, contiguous
      v16bf ahi, alo;
      ((v8bf*)&ahi)[0] = *(const v8bf*)(arhi + ak);
      ((v8bf*)&ahi)[1] = *(const v8bf*)(arhi + ak + 8);
      ((v8bf*)&alo)[0] = *(const v8bf*)(arlo + ak);
      ((v8bf*)&alo)[1] = *(const v8bf*)(arlo + ak + 8);
#pragma unroll
      for (int c = 0; c < 4; ++c) {
        v16bf bhi, blo;
        ((v8bf*)&bhi)[0] = *(const v8bf*)(brhi[c] + bk);
        ((v8bf*)&bhi)[1] = *(const v8bf*)(brhi[c] + bk + 8);
        ((v8bf*)&blo)[0] = *(const v8bf*)(brlo[c] + bk);
        ((v8bf*)&blo)[1] = *(const v8bf*)(brlo[c] + bk + 8);
        acc[c] = wmma_bf16(ahi, bhi, acc[c]);
        acc[c] = wmma_bf16(ahi, blo, acc[c]);
        acc[c] = wmma_bf16(alo, bhi, acc[c]);
      }
    }
#pragma unroll
    for (int c = 0; c < 4; ++c) {
      const int   j  = jbase + c * 16 + l16;
      const float cn = cnorm[j];
#pragma unroll
      for (int r = 0; r < 8; ++r) {
        float d = cn - 2.0f * acc[c][r];
        if (d < minv[r] || (d == minv[r] && j < mini[r])) { minv[r] = d; mini[r] = j; }
      }
    }
  }
  // butterfly min within each 16-lane half (same 8 rows, different cols)
#pragma unroll
  for (int m = 1; m < 16; m <<= 1) {
#pragma unroll
    for (int r = 0; r < 8; ++r) {
      float ov = __shfl_xor(minv[r], m, 32);
      int   oi = __shfl_xor(mini[r], m, 32);
      if (ov < minv[r] || (ov == minv[r] && oi < mini[r])) { minv[r] = ov; mini[r] = oi; }
    }
  }
  if (l16 == 0) {
#pragma unroll
    for (int r = 0; r < 8; ++r) {
      sval[wave][hf * 8 + r] = minv[r];
      sidx[wave][hf * 8 + r] = mini[r];
    }
  }
  __syncthreads();
  if (threadIdx.x < 16) {
    int row = threadIdx.x;
    float bv = sval[0][row];
    int   bi = sidx[0][row];
#pragma unroll
    for (int w = 1; w < 8; ++w) {
      float v  = sval[w][row];
      int   i2 = sidx[w][row];
      if (v < bv || (v == bv && i2 < bi)) { bv = v; bi = i2; }
    }
    minidx[row0 + row] = bi;
  }
}

// --------------------------------------------- scatter winners: freq & sums
__global__ __launch_bounds__(256) void scatter_kernel(const float* __restrict__ E,
                                                      const int* __restrict__ minidx,
                                                      float* __restrict__ freq,
                                                      float* __restrict__ sums) {
  const int i = blockIdx.x;
  const int j = minidx[i];
  if (threadIdx.x == 0) atomicAdd(&freq[j], 1.0f);
  for (int k = threadIdx.x; k < DIM; k += 256)
    atomicAdd(&sums[(size_t)j * DIM + k], E[(size_t)i * DIM + k]);
}

// ------------------------------- avg^T split bf16: avgT[d][j] = avg[j][d] split
__global__ __launch_bounds__(256) void avgT_kernel(const float* __restrict__ sums,
                                                   const float* __restrict__ freq,
                                                   const float* __restrict__ C,
                                                   __bf16* __restrict__ avgThi,
                                                   __bf16* __restrict__ avgTlo) {
  size_t idx = (size_t)blockIdx.x * 256 + threadIdx.x;
  if (idx >= (size_t)COMPS * DIM) return;
  int j = (int)(idx / DIM), k = (int)(idx % DIM);
  float f = freq[j];
  float a = (f > 0.5f) ? sums[idx] / f : C[idx];
  __bf16 h = (__bf16)a;
  avgThi[(size_t)k * COMPS + j] = h;
  avgTlo[(size_t)k * COMPS + j] = (__bf16)(a - (float)h);
}

// ------------- smoothing: new[i,d] = used ? (W@avg)[i,d]/wsum[i] : c ; + delta
// block = 16 comp rows x 768 dims; wave w owns 6 d-tiles (96 dims).
// W row-sums are fused into the tile build: grid_dists is read exactly once.
__global__ __launch_bounds__(256) void smooth_kernel(const float* __restrict__ GD,
                                                     const float* __restrict__ sigma,
                                                     const __bf16* __restrict__ avgThi,
                                                     const __bf16* __restrict__ avgTlo,
                                                     const float* __restrict__ freq,
                                                     const float* __restrict__ C,
                                                     float* __restrict__ out) {
  __shared__ __bf16 Whi[16 * 32];
  __shared__ __bf16 Wlo[16 * 32];
  __shared__ float wred[256];
  __shared__ float wsum_s[16];
  __shared__ float rowsq[8][16];
  __shared__ float rowdelta[16];

  const int lane = threadIdx.x & 31;
  const int wave = threadIdx.x >> 5;
  const int l16  = lane & 15;
  const int hf   = lane >> 4;
  const int i0   = blockIdx.x * 16;
  const float sg = sigma[0];
  const float inv2s2 = 1.0f / (2.0f * sg * sg);

  v8f acc[6] = {};
  float wpart = 0.0f;  // partial of sum_j W[row, j] for row = tid/16

  for (int jt = 0; jt < 128; ++jt) {
    const int j0 = jt * 32;
    // build 16x32 W tile (split bf16) cooperatively: 512 entries / 256 threads.
    // thread tid's two entries e=2*tid, 2*tid+1 are both in row tid/16.
#pragma unroll
    for (int q = 0; q < 2; ++q) {
      int e  = (int)threadIdx.x * 2 + q;
      int rr = e >> 5, cc = e & 31;
      float w = __expf(-GD[(size_t)(i0 + rr) * COMPS + j0 + cc] * inv2s2);
      wpart += w;
      __bf16 h = (__bf16)w;
      Whi[e] = h;
      Wlo[e] = (__bf16)(w - (float)h);
    }
    __syncthreads();
    v16bf ahi, alo;
    const int abase = l16 * 32 + hf * 16;
    ((v8bf*)&ahi)[0] = *(const v8bf*)(&Whi[abase]);
    ((v8bf*)&ahi)[1] = *(const v8bf*)(&Whi[abase + 8]);
    ((v8bf*)&alo)[0] = *(const v8bf*)(&Wlo[abase]);
    ((v8bf*)&alo)[1] = *(const v8bf*)(&Wlo[abase + 8]);
#pragma unroll
    for (int t = 0; t < 6; ++t) {
      const int d = wave * 96 + t * 16 + l16;
      const __bf16* bhp = avgThi + (size_t)d * COMPS + j0 + hf * 16;
      const __bf16* blp = avgTlo + (size_t)d * COMPS + j0 + hf * 16;
      v16bf bhi, blo;
      ((v8bf*)&bhi)[0] = *(const v8bf*)(bhp);
      ((v8bf*)&bhi)[1] = *(const v8bf*)(bhp + 8);
      ((v8bf*)&blo)[0] = *(const v8bf*)(blp);
      ((v8bf*)&blo)[1] = *(const v8bf*)(blp + 8);
      acc[t] = wmma_bf16(ahi, bhi, acc[t]);
      acc[t] = wmma_bf16(ahi, blo, acc[t]);
      acc[t] = wmma_bf16(alo, bhi, acc[t]);
    }
    __syncthreads();
  }

  // reduce W row sums: threads 16k..16k+15 all hold partials of row k
  wred[threadIdx.x] = wpart;
  __syncthreads();
#pragma unroll
  for (int m = 8; m > 0; m >>= 1) {
    if ((threadIdx.x & 15) < (unsigned)m) wred[threadIdx.x] += wred[threadIdx.x + m];
    __syncthreads();
  }
  if ((threadIdx.x & 15) == 0) wsum_s[threadIdx.x >> 4] = wred[threadIdx.x];
  __syncthreads();

  // finalize: divide by wsum, blend with old contents, write out, delta partials
  float rsq[8];
#pragma unroll
  for (int r = 0; r < 8; ++r) rsq[r] = 0.0f;
#pragma unroll
  for (int r = 0; r < 8; ++r) {
    const int   row  = hf * 8 + r;
    const int   i    = i0 + row;
    const float f    = freq[i];
    const float invw = 1.0f / wsum_s[row];
    const bool  used = f > 0.5f;
#pragma unroll
    for (int t = 0; t < 6; ++t) {
      const int   d   = wave * 96 + t * 16 + l16;
      const float c   = C[(size_t)i * DIM + d];
      const float val = used ? acc[t][r] * invw : c;
      out[1 + (size_t)i * DIM + d] = val;
      const float dd = val - c;
      rsq[r] += dd * dd;
    }
  }
#pragma unroll
  for (int m = 1; m < 16; m <<= 1) {
#pragma unroll
    for (int r = 0; r < 8; ++r) rsq[r] += __shfl_xor(rsq[r], m, 32);
  }
  if (l16 == 0) {
#pragma unroll
    for (int r = 0; r < 8; ++r) rowsq[wave][hf * 8 + r] = rsq[r];
  }
  __syncthreads();
  if (threadIdx.x < 16) {
    float s = 0.0f;
#pragma unroll
    for (int w = 0; w < 8; ++w) s += rowsq[w][threadIdx.x];
    rowdelta[threadIdx.x] = sqrtf(s);
  }
  __syncthreads();
  if (threadIdx.x == 0) {
    float s = 0.0f;
#pragma unroll
    for (int r = 0; r < 16; ++r) s += rowdelta[r];
    atomicAdd(out, s);
  }
}

// ---------------------------------------------------------------------- launch
extern "C" void kernel_launch(void* const* d_in, const int* in_sizes, int n_in,
                              void* d_out, int out_size, void* d_ws, size_t ws_size,
                              hipStream_t stream) {
  (void)in_sizes; (void)n_in; (void)out_size; (void)ws_size;
  const float* E   = (const float*)d_in[0];
  const float* C   = (const float*)d_in[1];
  const float* GD  = (const float*)d_in[2];
  const float* SIG = (const float*)d_in[3];
  float* out = (float*)d_out;

  char*  ws  = (char*)d_ws;
  size_t off = 0;
  auto take = [&](size_t bytes) -> char* {
    char* p = ws + off;
    off += (bytes + 255) & ~(size_t)255;
    return p;
  };
  __bf16* Ehi    = (__bf16*)take((size_t)NBATCH * DIM * 2);
  __bf16* Elo    = (__bf16*)take((size_t)NBATCH * DIM * 2);
  __bf16* Chi    = (__bf16*)take((size_t)COMPS * DIM * 2);
  __bf16* Clo    = (__bf16*)take((size_t)COMPS * DIM * 2);
  __bf16* avgThi = (__bf16*)take((size_t)DIM * COMPS * 2);
  __bf16* avgTlo = (__bf16*)take((size_t)DIM * COMPS * 2);
  float*  d_sums = (float*)take((size_t)COMPS * DIM * 4);
  float*  d_cn   = (float*)take((size_t)COMPS * 4);
  float*  d_fr   = (float*)take((size_t)COMPS * 4);
  int*    d_idx  = (int*)take((size_t)NBATCH * 4);

  zero_kernel<<<12288, 256, 0, stream>>>(d_fr, d_sums, out);
  split_kernel<<<(NBATCH * DIM) / 256, 256, 0, stream>>>(E, Ehi, Elo, (size_t)NBATCH * DIM);
  split_kernel<<<(COMPS * DIM) / 256, 256, 0, stream>>>(C, Chi, Clo, (size_t)COMPS * DIM);
  cnorm_kernel<<<COMPS, 256, 0, stream>>>(C, d_cn);
  bmu_kernel<<<NBATCH / 16, 256, 0, stream>>>(Ehi, Elo, Chi, Clo, d_cn, d_idx);
  scatter_kernel<<<NBATCH, 256, 0, stream>>>(E, d_idx, d_fr, d_sums);
  avgT_kernel<<<(COMPS * DIM) / 256, 256, 0, stream>>>(d_sums, d_fr, C, avgThi, avgTlo);
  smooth_kernel<<<COMPS / 16, 256, 0, stream>>>(GD, SIG, avgThi, avgTlo, d_fr, C, out);
}
